// UniGNNProcessorBoth_56384330662521
// MI455X (gfx1250) — compile-verified, matrix operand order checked by reading, not run on hardware
//
#include <hip/hip_runtime.h>
#include <hip/hip_bf16.h>

typedef __bf16 bf16;
typedef __attribute__((ext_vector_type(16))) __bf16 v16bf;
typedef __attribute__((ext_vector_type(8)))  __bf16 v8bf;
typedef __attribute__((ext_vector_type(4)))  __bf16 v4bf;
typedef __attribute__((ext_vector_type(8)))  float  v8f;

#define DEV_INLINE __device__ __forceinline__

union ABFrag { v16bf v; v8bf h[2]; };

DEV_INLINE v8f wmma_bf16(v16bf a, v16bf b, v8f c) {
  // D = A(16x32 bf16) * B(32x16 bf16) + C(16x16 f32)
  return __builtin_amdgcn_wmma_f32_16x16x32_bf16(false, a, false, b, (short)0, c,
                                                 false, false);
}

// Two 16x16 output tiles (rows rA and rA+16) sharing one B fragment per K-step.
DEV_INLINE void gemm2_lds(const bf16 (&xb)[32][128], const bf16* __restrict__ Wt,
                          int col, int rA, int khalf, v8f& o0, v8f& o1) {
  v8f acc0 = {}, acc1 = {};
  const bf16* brow = Wt + (size_t)col * 128;
#pragma unroll
  for (int kt = 0; kt < 4; ++kt) {
    const int k0 = kt * 32 + khalf * 8;
    ABFrag b, a0, a1;
    b.h[0]  = *(const v8bf*)(brow + k0);
    b.h[1]  = *(const v8bf*)(brow + k0 + 16);
    a0.h[0] = *(const v8bf*)&xb[rA][k0];
    a0.h[1] = *(const v8bf*)&xb[rA][k0 + 16];
    a1.h[0] = *(const v8bf*)&xb[rA + 16][k0];
    a1.h[1] = *(const v8bf*)&xb[rA + 16][k0 + 16];
    acc0 = wmma_bf16(a0.v, b.v, acc0);
    acc1 = wmma_bf16(a1.v, b.v, acc1);
  }
  o0 = acc0;
  o1 = acc1;
}

// ---------------------------------------------------------------------------
// Fused 3-layer MLP + LayerNorm + residual. Block = 32 rows x 128 cols,
// 8 waves, each wave owns one 16-col N-tile and two 16-row M-tiles.
// A tile staged to LDS via gfx1250 async copy (ASYNCcnt).
// ---------------------------------------------------------------------------
template <int IN>
__global__ __launch_bounds__(256) void mlp_ln_kernel(
    const bf16* __restrict__ Ain, const bf16* __restrict__ Wt,
    const float* __restrict__ prm, float* __restrict__ resid_out, long rows) {
  constexpr int KT0    = IN / 32;
  constexpr int OFF_B0 = 256 + IN * 128;
  constexpr int OFF_B1 = OFF_B0 + 128 + 16384;
  constexpr int OFF_B2 = OFF_B1 + 128 + 16384;
  constexpr int CHUNKS = 32 * IN / 8;  // 16-byte chunks in the A tile

  __shared__ __align__(16) bf16  Atile[32][IN];
  __shared__ __align__(16) bf16  xbuf[2][32][128];
  __shared__ __align__(16) float x3[32][128];
  __shared__ float redS[32][8];
  __shared__ float redQ[32][8];
  __shared__ float mur[32][2];

  const int t     = threadIdx.x;
  const int wave  = t >> 5;
  const int lid   = t & 31;
  const int col   = (wave << 4) | (lid & 15);  // output column 0..127
  const int khalf = lid >> 4;                  // K-half select
  const int rA    = lid & 15;                  // A-row within 16-row tile
  const long m0   = (long)blockIdx.x * 32;

  // ---- stage A tile (32 x IN bf16) into LDS with async copies ----
  {
    const bf16* gsrc = Ain + (size_t)m0 * IN;
    const unsigned lbase = (unsigned)(uintptr_t)&Atile[0][0];
#pragma unroll
    for (int i = t; i < CHUNKS; i += 256) {
      const bf16* g = gsrc + (size_t)i * 8;
      const unsigned l = lbase + i * 16;
      asm volatile("global_load_async_to_lds_b128 %0, %1, off"
                   :: "v"(l), "v"(g) : "memory");
    }
    asm volatile("s_wait_asynccnt 0x0" ::: "memory");
  }
  __syncthreads();

  // ---- layer 1: A from LDS, B from global (L2-resident weights) ----
  {
    v8f acc0 = {}, acc1 = {};
    const bf16* brow = Wt + (size_t)col * IN;
#pragma unroll
    for (int kt = 0; kt < KT0; ++kt) {
      const int k0 = kt * 32 + khalf * 8;
      ABFrag b, a0, a1;
      b.h[0]  = *(const v8bf*)(brow + k0);
      b.h[1]  = *(const v8bf*)(brow + k0 + 16);
      a0.h[0] = *(const v8bf*)&Atile[rA][k0];
      a0.h[1] = *(const v8bf*)&Atile[rA][k0 + 16];
      a1.h[0] = *(const v8bf*)&Atile[rA + 16][k0];
      a1.h[1] = *(const v8bf*)&Atile[rA + 16][k0 + 16];
      acc0 = wmma_bf16(a0.v, b.v, acc0);
      acc1 = wmma_bf16(a1.v, b.v, acc1);
    }
    const float bb = prm[OFF_B0 + col];
#pragma unroll
    for (int j = 0; j < 8; ++j) {
      float x = acc0[j] + bb; x = x > 0.f ? x : 0.f;
      xbuf[0][(khalf << 3) + j][col] = (bf16)x;
      float y = acc1[j] + bb; y = y > 0.f ? y : 0.f;
      xbuf[0][16 + (khalf << 3) + j][col] = (bf16)y;
    }
  }
  __syncthreads();

  // ---- layer 2 ----
  {
    v8f acc0, acc1;
    gemm2_lds(xbuf[0], Wt + (size_t)IN * 128, col, rA, khalf, acc0, acc1);
    const float bb = prm[OFF_B1 + col];
#pragma unroll
    for (int j = 0; j < 8; ++j) {
      float x = acc0[j] + bb; x = x > 0.f ? x : 0.f;
      xbuf[1][(khalf << 3) + j][col] = (bf16)x;
      float y = acc1[j] + bb; y = y > 0.f ? y : 0.f;
      xbuf[1][16 + (khalf << 3) + j][col] = (bf16)y;
    }
  }
  __syncthreads();

  // ---- layer 3 (no relu) -> fp32 LDS ----
  {
    v8f acc0, acc1;
    gemm2_lds(xbuf[1], Wt + (size_t)IN * 128 + 16384, col, rA, khalf, acc0, acc1);
    const float bb = prm[OFF_B2 + col];
#pragma unroll
    for (int j = 0; j < 8; ++j) {
      x3[(khalf << 3) + j][col]      = acc0[j] + bb;
      x3[16 + (khalf << 3) + j][col] = acc1[j] + bb;
    }
  }
  __syncthreads();

  // ---- LayerNorm: 32 teams of 8 threads ----
  {
    const int row = t >> 3, mem = t & 7;
    float s = 0.f, q = 0.f;
#pragma unroll
    for (int k = 0; k < 16; ++k) {
      const float v = x3[row][mem * 16 + k];
      s += v;
      q += v * v;
    }
    redS[row][mem] = s;
    redQ[row][mem] = q;
  }
  __syncthreads();
  if ((t & 7) == 0) {
    const int row = t >> 3;
    float S = 0.f, Q = 0.f;
#pragma unroll
    for (int k = 0; k < 8; ++k) { S += redS[row][k]; Q += redQ[row][k]; }
    const float mu  = S * (1.f / 128.f);
    const float var = Q * (1.f / 128.f) - mu * mu;
    mur[row][0] = mu;
    mur[row][1] = rsqrtf(var + 1e-5f);
  }
  __syncthreads();

  // ---- gamma/beta + residual, row-guarded stores ----
#pragma unroll
  for (int i = 0; i < 16; ++i) {
    const int e = t + i * 256;  // 0..4095
    const int r = e >> 7, c = e & 127;
    const long gr = m0 + r;
    if (gr < rows) {
      const float v = (x3[r][c] - mur[r][0]) * mur[r][1] * prm[128 + c] + prm[c];
      resid_out[gr * 128 + c] += v;
    }
  }
}

// ---------------------------------------------------------------------------
// Data-prep / scatter kernels (float4 vectorized)
// ---------------------------------------------------------------------------
__global__ void convert_w_kernel(const float* __restrict__ W, bf16* __restrict__ Wt,
                                 int in_dim) {
  const int idx = blockIdx.x * 256 + threadIdx.x;  // over in_dim*128
  if (idx >= in_dim * 128) return;
  const int i = idx >> 7, o = idx & 127;           // W[in][128] -> Wt[128][in]
  Wt[(size_t)o * in_dim + i] = (bf16)W[(size_t)i * 128 + o];
}

DEV_INLINE v4bf cvt4(float4 f) {
  v4bf r;
  r[0] = (bf16)f.x; r[1] = (bf16)f.y; r[2] = (bf16)f.z; r[3] = (bf16)f.w;
  return r;
}

__global__ void build_graph_edge_in(const float* __restrict__ xg,
                                    const float* __restrict__ efg,
                                    const int* __restrict__ vg,
                                    bf16* __restrict__ out, int E) {
  const int idx = blockIdx.x * 256 + threadIdx.x;  // over E*32
  if (idx >= E * 32) return;
  const int e = idx >> 5, c4 = (idx & 31) << 2;
  const int v0 = vg[2 * e], v1 = vg[2 * e + 1];
  bf16* o = out + (size_t)e * 384;
  *(v4bf*)(o + c4)       = cvt4(*(const float4*)(xg + (size_t)v0 * 128 + c4));
  *(v4bf*)(o + 128 + c4) = cvt4(*(const float4*)(xg + (size_t)v1 * 128 + c4));
  *(v4bf*)(o + 256 + c4) = cvt4(*(const float4*)(efg + (size_t)e * 128 + c4));
}

__global__ void build_hyper_edge_in(const float* __restrict__ xh,
                                    const float* __restrict__ efh,
                                    const int* __restrict__ vh,
                                    const int* __restrict__ he,
                                    bf16* __restrict__ out, int E, int nnz) {
  const int idx = blockIdx.x * 256 + threadIdx.x;  // over E*32
  if (idx >= E * 32) return;
  const int e = idx >> 5, c4 = (idx & 31) << 2;
  int lo = 0, hi = nnz;  // lower_bound(e)
  while (lo < hi) { const int m = (lo + hi) >> 1; if (he[m] < e) lo = m + 1; else hi = m; }
  const int start = lo;
  hi = nnz;              // lower_bound(e+1)
  while (lo < hi) { const int m = (lo + hi) >> 1; if (he[m] < e + 1) lo = m + 1; else hi = m; }
  float4 s = make_float4(0.f, 0.f, 0.f, 0.f);
  for (int i = start; i < lo; ++i) {
    const float4 v = *(const float4*)(xh + (size_t)vh[i] * 128 + c4);
    s.x += v.x; s.y += v.y; s.z += v.z; s.w += v.w;
  }
  bf16* o = out + (size_t)e * 256;
  *(v4bf*)(o + c4)       = cvt4(s);
  *(v4bf*)(o + 128 + c4) = cvt4(*(const float4*)(efh + (size_t)e * 128 + c4));
}

__global__ void zero_f32(float* __restrict__ p, int n4) {
  const int i = blockIdx.x * 256 + threadIdx.x;
  if (i < n4) *(float4*)(p + 4 * (size_t)i) = make_float4(0.f, 0.f, 0.f, 0.f);
}

__global__ void scatter_add_graph(const float* __restrict__ ef,
                                  const int* __restrict__ vg,
                                  float* __restrict__ Xv, int nnz) {
  const int idx = blockIdx.x * 256 + threadIdx.x;  // over nnz*32
  if (idx >= nnz * 32) return;
  const int i = idx >> 5, c4 = (idx & 31) << 2;
  const float4 v = *(const float4*)(ef + (size_t)(i >> 1) * 128 + c4);
  float* dst = Xv + (size_t)vg[i] * 128 + c4;
  atomicAdd(dst + 0, v.x);
  atomicAdd(dst + 1, v.y);
  atomicAdd(dst + 2, v.z);
  atomicAdd(dst + 3, v.w);
}

__global__ void scatter_add_hyper(const float* __restrict__ ef,
                                  const int* __restrict__ vh,
                                  const int* __restrict__ he,
                                  float* __restrict__ Xv, int nnz) {
  const int idx = blockIdx.x * 256 + threadIdx.x;  // over nnz*32
  if (idx >= nnz * 32) return;
  const int i = idx >> 5, c4 = (idx & 31) << 2;
  const float4 v = *(const float4*)(ef + (size_t)he[i] * 128 + c4);
  float* dst = Xv + (size_t)vh[i] * 128 + c4;
  atomicAdd(dst + 0, v.x);
  atomicAdd(dst + 1, v.y);
  atomicAdd(dst + 2, v.z);
  atomicAdd(dst + 3, v.w);
}

__global__ void build_node_in(const float* __restrict__ Xv,
                              const float* __restrict__ x,
                              bf16* __restrict__ out, int N) {
  const int idx = blockIdx.x * 256 + threadIdx.x;  // over N*32
  if (idx >= N * 32) return;
  const int n = idx >> 5, c4 = (idx & 31) << 2;
  bf16* o = out + (size_t)n * 256;
  *(v4bf*)(o + c4)       = cvt4(*(const float4*)(Xv + (size_t)n * 128 + c4));
  *(v4bf*)(o + 128 + c4) = cvt4(*(const float4*)(x + (size_t)n * 128 + c4));
}

__global__ void add2_kernel(const float* __restrict__ a, const float* __restrict__ b,
                            float* __restrict__ o, int n4) {
  const int i = blockIdx.x * 256 + threadIdx.x;
  if (i >= n4) return;
  const float4 x = *(const float4*)(a + 4 * (size_t)i);
  const float4 y = *(const float4*)(b + 4 * (size_t)i);
  *(float4*)(o + 4 * (size_t)i) = make_float4(x.x + y.x, x.y + y.y, x.z + y.z, x.w + y.w);
}

// ---------------------------------------------------------------------------
extern "C" void kernel_launch(void* const* d_in, const int* in_sizes, int n_in,
                              void* d_out, int out_size, void* d_ws, size_t ws_size,
                              hipStream_t stream) {
  (void)in_sizes; (void)n_in; (void)out_size; (void)ws_size;
  const int N = 30000, E = 150000, NHE = 40000, NNZ = 300000, D = 128;

  const float* in_xg  = (const float*)d_in[0];
  const int*   vg     = (const int*)d_in[1];
  // d_in[2] = hyperedges_g (repeat(arange(E),2)) -> implicit, unused
  const float* in_efg = (const float*)d_in[3];
  const float* in_xh  = (const float*)d_in[4];
  const int*   vh     = (const int*)d_in[5];
  const int*   heh    = (const int*)d_in[6];
  const float* in_efh = (const float*)d_in[7];
  const float* prm    = (const float*)d_in[8];  // pytree-flat fp32 params

  // MLP order (jax dict keys sorted): gnn[s].edge, gnn[s].node, then hnn.
  static const int in_dims[12] = {384, 256, 384, 256, 384, 256,
                                  256, 256, 256, 256, 256, 256};
  size_t poff[12], woff[12], pc = 0, wc = 0;
  for (int m = 0; m < 12; ++m) {
    poff[m] = pc; woff[m] = wc;
    // beta(128)+gamma(128)+W0(in*128)+b0(128)+W1(16384)+b1(128)+W2(16384)+b2(128)
    pc += 256 + (size_t)in_dims[m] * 128 + 128 + 16384 + 128 + 16384 + 128;
    wc += (size_t)in_dims[m] * 128 + 16384 + 16384;  // bf16 transposed weights
  }

  // workspace carve (256B aligned)
  char* w = (char*)d_ws;
  auto carve = [&](size_t bytes) -> char* {
    char* p = w;
    w += (bytes + 255) & ~(size_t)255;
    return p;
  };
  bf16*  Wts   = (bf16*)carve(wc * sizeof(bf16));
  float* xg    = (float*)carve((size_t)N * D * 4);
  float* efg   = (float*)carve((size_t)E * D * 4);
  float* xh    = (float*)carve((size_t)N * D * 4);
  float* efh   = (float*)carve((size_t)NHE * D * 4);
  float* Xv    = (float*)carve((size_t)N * D * 4);
  bf16*  stage = (bf16*)carve((size_t)(E + 32) * 384 * sizeof(bf16));  // +tile pad

  hipMemcpyAsync(xg,  in_xg,  (size_t)N * D * 4,   hipMemcpyDeviceToDevice, stream);
  hipMemcpyAsync(efg, in_efg, (size_t)E * D * 4,   hipMemcpyDeviceToDevice, stream);
  hipMemcpyAsync(xh,  in_xh,  (size_t)N * D * 4,   hipMemcpyDeviceToDevice, stream);
  hipMemcpyAsync(efh, in_efh, (size_t)NHE * D * 4, hipMemcpyDeviceToDevice, stream);

  // transpose + bf16-convert all weights (L2-resident afterwards)
  for (int m = 0; m < 12; ++m) {
    const float* p = prm + poff[m];
    const int in = in_dims[m];
    bf16* wt = Wts + woff[m];
    convert_w_kernel<<<(in * 128 + 255) / 256, 256, 0, stream>>>(p + 256, wt, in);
    convert_w_kernel<<<(128 * 128 + 255) / 256, 256, 0, stream>>>(
        p + 256 + (size_t)in * 128 + 128, wt + (size_t)in * 128, 128);
    convert_w_kernel<<<(128 * 128 + 255) / 256, 256, 0, stream>>>(
        p + 256 + (size_t)in * 128 + 128 + 16384 + 128,
        wt + (size_t)in * 128 + 16384, 128);
  }

  // ---- 3 graph steps ----
  for (int s = 0; s < 3; ++s) {
    const int me = 2 * s, mn = 2 * s + 1;
    build_graph_edge_in<<<E * 32 / 256, 256, 0, stream>>>(xg, efg, vg, stage, E);
    mlp_ln_kernel<384><<<(E + 31) / 32, 256, 0, stream>>>(
        stage, Wts + woff[me], prm + poff[me], efg, E);
    zero_f32<<<N * D / 4 / 256, 256, 0, stream>>>(Xv, N * D / 4);
    scatter_add_graph<<<2 * E * 32 / 256, 256, 0, stream>>>(efg, vg, Xv, 2 * E);
    build_node_in<<<N * 32 / 256, 256, 0, stream>>>(Xv, xg, stage, N);
    mlp_ln_kernel<256><<<(N + 31) / 32, 256, 0, stream>>>(
        stage, Wts + woff[mn], prm + poff[mn], xg, N);
  }

  // ---- 3 hypergraph steps ----
  for (int s = 0; s < 3; ++s) {
    const int me = 6 + 2 * s, mn = 7 + 2 * s;
    build_hyper_edge_in<<<NHE * 32 / 256, 256, 0, stream>>>(xh, efh, vh, heh,
                                                            stage, NHE, NNZ);
    mlp_ln_kernel<256><<<(NHE + 31) / 32, 256, 0, stream>>>(
        stage, Wts + woff[me], prm + poff[me], efh, NHE);
    zero_f32<<<N * D / 4 / 256, 256, 0, stream>>>(Xv, N * D / 4);
    scatter_add_hyper<<<NNZ * 32 / 256, 256, 0, stream>>>(efh, vh, heh, Xv, NNZ);
    build_node_in<<<N * 32 / 256, 256, 0, stream>>>(Xv, xh, stage, N);
    mlp_ln_kernel<256><<<(N + 31) / 32, 256, 0, stream>>>(
        stage, Wts + woff[mn], prm + poff[mn], xh, N);
  }

  // ---- outputs: [x_h + x_g | ef_g] ----
  add2_kernel<<<N * D / 4 / 256, 256, 0, stream>>>(xh, xg, (float*)d_out, N * D / 4);
  hipMemcpyAsync((float*)d_out + (size_t)N * D, efg, (size_t)E * D * 4,
                 hipMemcpyDeviceToDevice, stream);
}
